// ReLuPCA_13176959664583
// MI455X (gfx1250) — matrix operand order repaired and checked by reading, not compiled, not run on hardware
//
#include <hip/hip_runtime.h>

// ---------------------------------------------------------------------------
// ReLU-PCA 8-bit fake-quant pipeline for MI455X (gfx1250, wave32, WMMA).
//   x:(32,256,56,56) f32, u:(256,256) f32, clamp:(256) f32 -> out f32.
// Two 256x256x100352 GEMMs done in bf16 WMMA (v_wmma_f32_16x16x32_bf16),
// f32 accumulate. All big intermediates live in d_ws (L2-resident on MI455X:
// 192MB L2 vs ~155MB ws) so HBM traffic ~= read x + write out (~206MB).
// ---------------------------------------------------------------------------

typedef __attribute__((ext_vector_type(16))) __bf16 v16bf;
typedef __attribute__((ext_vector_type(8)))  float  v8f;

#define RP_C   256
#define RP_HW  3136        // 56*56
#define RP_N   32
#define RP_M   100352      // RP_N*RP_HW
#define RP_INVM (1.0f / 100352.0f)

// ---- bf16 helpers (raw u16 storage; RNE convert) --------------------------
__device__ __forceinline__ unsigned short rp_f2bf(float f) {
    unsigned u = __float_as_uint(f);
    unsigned r = 0x7FFFu + ((u >> 16) & 1u);
    return (unsigned short)((u + r) >> 16);
}
__device__ __forceinline__ float rp_bf2f(unsigned short h) {
    return __uint_as_float(((unsigned)h) << 16);
}

// ---- float atomic min/max via CAS -----------------------------------------
__device__ __forceinline__ void rp_atomicMinF(float* addr, float v) {
    unsigned* ua = reinterpret_cast<unsigned*>(addr);
    unsigned cur = *ua;
    while (__uint_as_float(cur) > v) {
        unsigned prev = atomicCAS(ua, cur, __float_as_uint(v));
        if (prev == cur) break;
        cur = prev;
    }
}
__device__ __forceinline__ void rp_atomicMaxF(float* addr, float v) {
    unsigned* ua = reinterpret_cast<unsigned*>(addr);
    unsigned cur = *ua;
    while (__uint_as_float(cur) < v) {
        unsigned prev = atomicCAS(ua, cur, __float_as_uint(v));
        if (prev == cur) break;
        cur = prev;
    }
}

// ---- WMMA fragment loads ---------------------------------------------------
// A (16x32 bf16, row-major source, ld=256): lane holds K = kb+lh*8 .. +7 and
// kb+lh*8+16 .. +23  -> two 16B loads.
__device__ __forceinline__ v16bf rp_ldA(const unsigned short* p) {
    union { v16bf v; uint4 u[2]; } f;
    f.u[0] = *(const uint4*)(p);
    f.u[1] = *(const uint4*)(p + 16);
    return f.v;
}
// B (32x16 bf16, stored K-innermost): lane holds K = kb+lh*8 .. +7 and
// kb+lh*8+16 .. +23 -> two 16B loads.
__device__ __forceinline__ v16bf rp_ldB(const unsigned short* p) {
    union { v16bf v; uint4 u[2]; } f;
    f.u[0] = *(const uint4*)(p);
    f.u[1] = *(const uint4*)(p + 16);
    return f.v;
}

// ---------------------------------------------------------------------------
// K0: init accumulators
__global__ void rp_init(float* chansum, float* dmin, float* dmax, float* rowsum) {
    int t = threadIdx.x;
    chansum[t] = 0.0f;
    rowsum[t]  = 0.0f;
    dmin[t]    =  3.402823e38f;
    dmax[t]    = -3.402823e38f;
}

// ---------------------------------------------------------------------------
// K1: relu(x) -> im_t (M x C, bf16, K-innermost for B-fragments) + channel sums.
// 32c x 32hw LDS tile transpose; coalesced 128B reads / 64B writes.
__global__ __launch_bounds__(256) void rp_relu_pack(
    const float* __restrict__ x, unsigned short* __restrict__ imt,
    float* __restrict__ chansum)
{
    __shared__ float tile[32][33];
    __shared__ float psum[256];
    const int hw0 = blockIdx.x * 32;
    const int c0  = blockIdx.y * 32;
    const int n   = blockIdx.z;
    const int col = threadIdx.x & 31;
    const int grp = threadIdx.x >> 5;           // 0..7

#pragma unroll
    for (int rr = 0; rr < 4; ++rr) {
        const int cl = grp + rr * 8;
        float v = x[((size_t)(n * RP_C + c0 + cl)) * RP_HW + hw0 + col];
        tile[cl][col] = fmaxf(v, 0.0f);
    }
    __syncthreads();

    float s = 0.0f;
#pragma unroll
    for (int rr = 0; rr < 4; ++rr) {
        const int hwl = grp + rr * 8;
        const float v = tile[col][hwl];
        const size_t m = (size_t)n * RP_HW + hw0 + hwl;
        imt[m * RP_C + c0 + col] = rp_f2bf(v);
        s += v;
    }
    psum[threadIdx.x] = s;
    __syncthreads();
    if (threadIdx.x < 32) {
        float t = 0.0f;
#pragma unroll
        for (int g = 0; g < 8; ++g) t += psum[threadIdx.x + 32 * g];
        atomicAdd(&chansum[c0 + threadIdx.x], t);
    }
}

// ---------------------------------------------------------------------------
// K2: mn[c], off[k] = sum_c u[c,k]*mn[c], and bf16 copies of u / u^T.
__global__ void rp_prep(const float* __restrict__ u,
                        const float* __restrict__ chansum,
                        float* __restrict__ mn, float* __restrict__ off,
                        unsigned short* __restrict__ uT,
                        unsigned short* __restrict__ uB)
{
    const int t = threadIdx.x;                  // 256 threads, 1 block
    mn[t] = chansum[t] * RP_INVM;
    __syncthreads();
    float o = 0.0f;
    for (int c = 0; c < RP_C; ++c) o += u[c * RP_C + t] * mn[c];
    off[t] = o;
    for (int c = 0; c < RP_C; ++c) uT[t * RP_C + c] = rp_f2bf(u[c * RP_C + t]); // uT[k][c]
    for (int k = 0; k < RP_C; ++k) uB[t * RP_C + k] = rp_f2bf(u[t * RP_C + k]); // u [c][k]
}

// ---------------------------------------------------------------------------
// K3: GEMM1  p = uT @ relu(x) - off ; clamp ; store pq (C x M f32) ; row min/max.
// 8 wave32 per block; wave w owns rows 32w..32w+31, block owns 64 m-columns.
__global__ __launch_bounds__(256) void rp_gemm1(
    const unsigned short* __restrict__ imt, const unsigned short* __restrict__ uT,
    const float* __restrict__ off, const float* __restrict__ clampv,
    float* __restrict__ pq, float* __restrict__ dmin, float* __restrict__ dmax)
{
    const int m0   = blockIdx.x * 64;
    const int wave = threadIdx.x >> 5;
    const int lane = threadIdx.x & 31;
    const int l16  = lane & 15;
    const int lh   = lane >> 4;

    v8f acc[2][4] = {};

#pragma unroll
    for (int ks = 0; ks < 8; ++ks) {
        const int kb = ks * 32;
        v16bf a[2], b[4];
#pragma unroll
        for (int rt = 0; rt < 2; ++rt) {
            const int row = wave * 32 + rt * 16 + l16;
            a[rt] = rp_ldA(uT + (size_t)row * RP_C + kb + lh * 8);
        }
#pragma unroll
        for (int ct = 0; ct < 4; ++ct) {
            const int m = m0 + ct * 16 + l16;
            const unsigned short* p = imt + (size_t)m * RP_C + kb + lh * 8;
            b[ct] = rp_ldB(p);
            if (ks < 7) __builtin_prefetch(p + 32, 0, 0);   // global_prefetch_b8
        }
#pragma unroll
        for (int rt = 0; rt < 2; ++rt)
#pragma unroll
            for (int ct = 0; ct < 4; ++ct)
                acc[rt][ct] = __builtin_amdgcn_wmma_f32_16x16x32_bf16(
                    false, a[rt], false, b[ct], (short)0, acc[rt][ct], false, false);
    }

#pragma unroll
    for (int rt = 0; rt < 2; ++rt) {
        const int kbase = wave * 32 + rt * 16 + lh * 8;
#pragma unroll
        for (int r = 0; r < 8; ++r) {
            const int k = kbase + r;
            const float ofk = off[k];
            const float ck  = clampv[k];
            float vmn = 3.402823e38f, vmx = -3.402823e38f;
#pragma unroll
            for (int ct = 0; ct < 4; ++ct) {
                float v = acc[rt][ct][r] - ofk;
                v = fminf(fmaxf(v, -ck), ck);
                pq[(size_t)k * RP_M + (m0 + ct * 16 + l16)] = v;   // 64B coalesced
                vmn = fminf(vmn, v);
                vmx = fmaxf(vmx, v);
            }
#pragma unroll
            for (int d = 1; d < 16; d <<= 1) {
                vmn = fminf(vmn, __shfl_xor(vmn, d, 32));
                vmx = fmaxf(vmx, __shfl_xor(vmx, d, 32));
            }
            if (l16 == 0) {
                rp_atomicMinF(&dmin[k], vmn);
                rp_atomicMaxF(&dmax[k], vmx);
            }
        }
    }
}

// ---------------------------------------------------------------------------
// K4: quant params per row; qs==0 encodes rng==0 passthrough.
__global__ void rp_qparams(const float* __restrict__ dmin, const float* __restrict__ dmax,
                           float* __restrict__ qs, float* __restrict__ qi,
                           float* __restrict__ dm)
{
    const int k = threadIdx.x;
    const float lo = dmin[k], hi = dmax[k];
    const float rng = hi - lo;
    if (rng == 0.0f) { qs[k] = 0.0f; qi[k] = 0.0f; }
    else             { qs[k] = 255.0f / rng; qi[k] = rng * (1.0f / 255.0f); }
    dm[k] = lo;
}

// ---------------------------------------------------------------------------
// K5: fake-quantize pq -> q_t (M x C bf16, K-innermost), accumulate rowsum(q).
__global__ __launch_bounds__(256) void rp_quant(
    const float* __restrict__ pq, const float* __restrict__ qs,
    const float* __restrict__ qi, const float* __restrict__ dm,
    unsigned short* __restrict__ qt, float* __restrict__ rowsum)
{
    __shared__ unsigned short ql[64][258];      // +2 pad: dodge bank conflicts
    __shared__ float sqs[256], sqi[256], sdm[256];
    const int t = threadIdx.x;
    sqs[t] = qs[t]; sqi[t] = qi[t]; sdm[t] = dm[t];
    __syncthreads();

    const int m0 = blockIdx.x * 64;
    const int ml = t & 63;
    const int kq = t >> 6;                      // 0..3
    for (int kb = 0; kb < RP_C; kb += 16) {
#pragma unroll
        for (int rr = 0; rr < 4; ++rr) {
            const int k = kb + kq * 4 + rr;
            const float v = pq[(size_t)k * RP_M + m0 + ml];   // 256B coalesced
            const float s = sqs[k];
            const float q = (s == 0.0f) ? v
                          : (rintf((v - sdm[k]) * s) * sqi[k] + sdm[k]);
            ql[ml][k] = rp_f2bf(q);
        }
    }
    __syncthreads();

    float rs = 0.0f;
    for (int mm = 0; mm < 64; ++mm) {
        const unsigned short h = ql[mm][t];
        rs += rp_bf2f(h);
        qt[((size_t)(m0 + mm)) * RP_C + t] = h;               // 512B coalesced
    }
    atomicAdd(&rowsum[t], rs);
}

// ---------------------------------------------------------------------------
// K6: bias[c] = mn[c] - (u @ rowsum_q)[c] / M
__global__ void rp_bias(const float* __restrict__ u, const float* __restrict__ rowsum,
                        const float* __restrict__ mn, float* __restrict__ bias)
{
    const int c = threadIdx.x;
    float s = 0.0f;
    for (int k = 0; k < RP_C; ++k) s += u[c * RP_C + k] * rowsum[k];
    bias[c] = mn[c] - s * RP_INVM;
}

// ---------------------------------------------------------------------------
// K7: GEMM2  out = u @ q + bias, scattered straight into NCHW layout.
__global__ __launch_bounds__(256) void rp_gemm2(
    const unsigned short* __restrict__ qt, const unsigned short* __restrict__ uB,
    const float* __restrict__ bias, float* __restrict__ out)
{
    const int m0   = blockIdx.x * 64;
    const int wave = threadIdx.x >> 5;
    const int lane = threadIdx.x & 31;
    const int l16  = lane & 15;
    const int lh   = lane >> 4;

    v8f acc[2][4] = {};

#pragma unroll
    for (int ks = 0; ks < 8; ++ks) {
        const int kb = ks * 32;
        v16bf a[2], b[4];
#pragma unroll
        for (int rt = 0; rt < 2; ++rt) {
            const int row = wave * 32 + rt * 16 + l16;        // output channel c
            a[rt] = rp_ldA(uB + (size_t)row * RP_C + kb + lh * 8);
        }
#pragma unroll
        for (int ct = 0; ct < 4; ++ct) {
            const int m = m0 + ct * 16 + l16;
            b[ct] = rp_ldB(qt + (size_t)m * RP_C + kb + lh * 8);
        }
#pragma unroll
        for (int rt = 0; rt < 2; ++rt)
#pragma unroll
            for (int ct = 0; ct < 4; ++ct)
                acc[rt][ct] = __builtin_amdgcn_wmma_f32_16x16x32_bf16(
                    false, a[rt], false, b[ct], (short)0, acc[rt][ct], false, false);
    }

#pragma unroll
    for (int rt = 0; rt < 2; ++rt) {
        const int cbase = wave * 32 + rt * 16 + lh * 8;
#pragma unroll
        for (int r = 0; r < 8; ++r) {
            const int c  = cbase + r;
            const float bc = bias[c];
#pragma unroll
            for (int ct = 0; ct < 4; ++ct) {
                const int m  = m0 + ct * 16 + l16;
                const int n  = m / RP_HW;                     // 16-tile never crosses n
                const int hw = m - n * RP_HW;
                out[((size_t)(n * RP_C + c)) * RP_HW + hw] = acc[rt][ct][r] + bc;
            }
        }
    }
}

// ---------------------------------------------------------------------------
extern "C" void kernel_launch(void* const* d_in, const int* in_sizes, int n_in,
                              void* d_out, int out_size, void* d_ws, size_t ws_size,
                              hipStream_t stream)
{
    (void)in_sizes; (void)n_in; (void)out_size; (void)ws_size;
    const float* x  = (const float*)d_in[0];
    const float* u  = (const float*)d_in[1];
    const float* cv = (const float*)d_in[2];
    float* out = (float*)d_out;

    char* ws = (char*)d_ws;
    size_t o = 0;
    auto carve = [&](size_t bytes) -> void* {
        void* p = ws + o;
        o += (bytes + 4095) & ~(size_t)4095;
        return p;
    };
    unsigned short* imt = (unsigned short*)carve((size_t)RP_M * RP_C * 2); // reused as q_t
    float*          pq  = (float*)carve((size_t)RP_C * RP_M * 4);
    unsigned short* uT  = (unsigned short*)carve((size_t)RP_C * RP_C * 2);
    unsigned short* uB  = (unsigned short*)carve((size_t)RP_C * RP_C * 2);
    float* chansum = (float*)carve(RP_C * 4);
    float* mn      = (float*)carve(RP_C * 4);
    float* off     = (float*)carve(RP_C * 4);
    float* dmin    = (float*)carve(RP_C * 4);
    float* dmax    = (float*)carve(RP_C * 4);
    float* qsv     = (float*)carve(RP_C * 4);
    float* qiv     = (float*)carve(RP_C * 4);
    float* dmv     = (float*)carve(RP_C * 4);
    float* rowsum  = (float*)carve(RP_C * 4);
    float* bias    = (float*)carve(RP_C * 4);
    unsigned short* qt = imt;   // im_t dead after GEMM1 -> reuse for q_t

    rp_init<<<1, 256, 0, stream>>>(chansum, dmin, dmax, rowsum);
    rp_relu_pack<<<dim3(RP_HW / 32, RP_C / 32, RP_N), 256, 0, stream>>>(x, imt, chansum);
    rp_prep<<<1, 256, 0, stream>>>(u, chansum, mn, off, uT, uB);
    rp_gemm1<<<RP_M / 64, 256, 0, stream>>>(imt, uT, off, cv, pq, dmin, dmax);
    rp_qparams<<<1, 256, 0, stream>>>(dmin, dmax, qsv, qiv, dmv);
    rp_quant<<<RP_M / 64, 256, 0, stream>>>(pq, qsv, qiv, dmv, qt, rowsum);
    rp_bias<<<1, 256, 0, stream>>>(u, rowsum, mn, bias);
    rp_gemm2<<<RP_M / 64, 256, 0, stream>>>(qt, uB, bias, out);
}
